// MambaMia2Mixer_41884521071067
// MI455X (gfx1250) — matrix operand order repaired and checked by reading, not compiled, not run on hardware
//
#include <hip/hip_runtime.h>
#include <hip/hip_bf16.h>
#include <math.h>

// ---------------- problem constants ----------------
constexpr int kB     = 2;
constexpr int kL     = 2048;
constexpr int kT     = kB * kL;          // 4096 tokens
constexpr int kHID   = 2048;
constexpr int kH     = 64;               // heads
constexpr int kP     = 64;               // head dim
constexpr int kG     = 8;                // groups
constexpr int kN     = 128;              // state dim
constexpr int kK     = 4;                // conv width
constexpr int kCS    = 256;              // chunk size
constexpr int kNC    = kL / kCS;         // 8 chunks per batch
constexpr int kINTER = kH * kP;          // 4096
constexpr int kCONVD = kINTER + 2 * kG * kN;   // 6144
constexpr int kPROJ  = kINTER + kCONVD + kH;   // 10304
constexpr int kGN    = kG * kN;          // 1024
constexpr float kEPS = 1e-5f;

// ---------------- vector types ----------------
typedef __attribute__((ext_vector_type(16))) __bf16 v16bf;
typedef __attribute__((ext_vector_type(8)))  __bf16 v8bf;
typedef __attribute__((ext_vector_type(8)))  float  v8f;
typedef __attribute__((ext_vector_type(4)))  int    v4i;

static __device__ __forceinline__ v8f vzero8() {
  v8f z = {0.f, 0.f, 0.f, 0.f, 0.f, 0.f, 0.f, 0.f};
  return z;
}

static __device__ __forceinline__ __bf16 f2bf(float f) {
  unsigned u = __builtin_bit_cast(unsigned, f);
  unsigned r = u + 0x7FFFu + ((u >> 16) & 1u);   // round-to-nearest-even
  unsigned short h = (unsigned short)(r >> 16);
  return __builtin_bit_cast(__bf16, h);
}
static __device__ __forceinline__ float bf2f(__bf16 b) {
  unsigned short h = __builtin_bit_cast(unsigned short, b);
  unsigned u = ((unsigned)h) << 16;
  return __builtin_bit_cast(float, u);
}

static __device__ __forceinline__ v16bf mk16(v8bf a, v8bf b) {
  v16bf r;
#pragma unroll
  for (int i = 0; i < 8; ++i) { r[i] = a[i]; r[8 + i] = b[i]; }
  return r;
}

// A-fragment (16xK=32 bf16). base points at element [row0][k0] of a row-major
// [rows][ld] buffer.  Lane holds row = lane%16; lanes 0-15 carry
// K 0..15, lanes 16-31 carry K 16..31 (matches load_b_frag).
static __device__ __forceinline__ v16bf load_a_frag(const __bf16* base, int ld, int lane) {
  const int r = lane & 15, half = lane >> 4;
  const __bf16* p = base + (size_t)r * ld;
  v8bf x0 = *(const v8bf*)(p + half * 16);
  v8bf x1 = *(const v8bf*)(p + half * 16 + 8);
  return mk16(x0, x1);
}

// B-fragment (K=32 x 16 bf16). base points at element [col0][k0] of a
// row-major [cols][ld] buffer holding B^T (K contiguous per column).
// Lane holds column = lane%16; lanes 0-15 carry K 0..15, lanes 16-31 K 16..31.
static __device__ __forceinline__ v16bf load_b_frag(const __bf16* base, int ld, int lane) {
  const int c = lane & 15, half = lane >> 4;
  const __bf16* p = base + (size_t)c * ld + half * 16;
  return mk16(*(const v8bf*)p, *(const v8bf*)(p + 8));
}

static __device__ __forceinline__ v8f wmma_bf16(v16bf a, v16bf b, v8f c) {
  return __builtin_amdgcn_wmma_f32_16x16x32_bf16(false, a, false, b, (short)0, c,
                                                 false, false);
}

// ---------------- CDNA5 async global->LDS copy (ASYNCcnt-tracked) ----------
#if defined(__gfx1250__) && __has_builtin(__builtin_amdgcn_global_load_async_to_lds_b128)
#define HAVE_ASYNC_LDS 1
static __device__ __forceinline__ void g2l_b128(const __bf16* g, __bf16* l) {
  __builtin_amdgcn_global_load_async_to_lds_b128(
      (__attribute__((address_space(1))) v4i*)g,
      (__attribute__((address_space(3))) v4i*)l, 0, 0);
}
static __device__ __forceinline__ void wait_async3() {
#if __has_builtin(__builtin_amdgcn_s_wait_asynccnt)
  __builtin_amdgcn_s_wait_asynccnt(3);
#else
  asm volatile("s_wait_asynccnt 0x3" ::: "memory");
#endif
}
static __device__ __forceinline__ void wait_async0() {
#if __has_builtin(__builtin_amdgcn_s_wait_asynccnt)
  __builtin_amdgcn_s_wait_asynccnt(0);
#else
  asm volatile("s_wait_asynccnt 0x0" ::: "memory");
#endif
}
#else
#define HAVE_ASYNC_LDS 0
#endif

// ---------------- fp32 -> bf16 convert ----------------
__global__ __launch_bounds__(256) void cvt_bf16_kernel(const float* __restrict__ in,
                                                       __bf16* __restrict__ out,
                                                       size_t n) {
  size_t i = (size_t)blockIdx.x * blockDim.x + threadIdx.x;
  size_t stride = (size_t)gridDim.x * blockDim.x;
  for (; i < n; i += stride) out[i] = f2bf(in[i]);
}

// ---------------- generic bf16 GEMM: C[M,N] = A[M,K] * Bw[N,K]^T ----------------
// Block tile 128(M) x 64(N), K-step 32, 8 waves, each wave 32x32 (2x2 WMMA tiles).
// Double-buffered async global->LDS staging on CDNA5; last K-step peeled so the
// steady-state loop is branch-free (issue next tile, wait ASYNCcnt<=3, compute).
__global__ __launch_bounds__(256) void gemm_bf16_nt(const __bf16* __restrict__ A,
                                                    const __bf16* __restrict__ Bw,
                                                    float* __restrict__ C,
                                                    int M, int N, int Kd,
                                                    int lda, int ldb, int ldc) {
  __shared__ __align__(16) __bf16 As[2][128][40];
  __shared__ __align__(16) __bf16 Bs[2][64][40];
  const int tid  = threadIdx.x;
  const int lane = tid & 31;
  const int wid  = tid >> 5;
  const int m0 = blockIdx.y * 128;
  const int n0 = blockIdx.x * 64;
  const int wm = (wid & 3) * 32;
  const int wn = (wid >> 2) * 32;

  // per-thread staging assignment
  const int arow = tid >> 1, akof = (tid & 1) * 16;   // A: 128x32, 16 elems/thread
  const int brow = tid >> 2, bkof = (tid & 3) * 8;    // B: 64x32,  8 elems/thread
  const __bf16* gA = A  + (size_t)(m0 + arow) * lda + akof;
  const __bf16* gB = Bw + (size_t)(n0 + brow) * ldb + bkof;

  v8f acc[2][2];
#pragma unroll
  for (int i = 0; i < 2; ++i)
#pragma unroll
    for (int j = 0; j < 2; ++j) acc[i][j] = vzero8();

#if HAVE_ASYNC_LDS
  // ---- async, double-buffered, branch-free steady state ----
  g2l_b128(gA, &As[0][arow][akof]);
  g2l_b128(gA + 8, &As[0][arow][akof + 8]);
  g2l_b128(gB, &Bs[0][brow][bkof]);

  int k0 = 0;
  for (; k0 + 32 < Kd; k0 += 32) {
    const int buf = (k0 >> 5) & 1;
    const int kn = k0 + 32;
    g2l_b128(gA + kn, &As[buf ^ 1][arow][akof]);
    g2l_b128(gA + kn + 8, &As[buf ^ 1][arow][akof + 8]);
    g2l_b128(gB + kn, &Bs[buf ^ 1][brow][bkof]);
    wait_async3();            // current tile's 3 ops done (in-order completion)
    __syncthreads();

    v16bf af[2], bfr[2];
    af[0]  = load_a_frag(&As[buf][wm][0],      40, lane);
    af[1]  = load_a_frag(&As[buf][wm + 16][0], 40, lane);
    bfr[0] = load_b_frag(&Bs[buf][wn][0],      40, lane);
    bfr[1] = load_b_frag(&Bs[buf][wn + 16][0], 40, lane);
#pragma unroll
    for (int ti = 0; ti < 2; ++ti)
#pragma unroll
      for (int tj = 0; tj < 2; ++tj)
        acc[ti][tj] = wmma_bf16(af[ti], bfr[tj], acc[ti][tj]);
    __syncthreads();
  }
  {  // peeled final K-step
    const int buf = (k0 >> 5) & 1;
    wait_async0();
    __syncthreads();
    v16bf af[2], bfr[2];
    af[0]  = load_a_frag(&As[buf][wm][0],      40, lane);
    af[1]  = load_a_frag(&As[buf][wm + 16][0], 40, lane);
    bfr[0] = load_b_frag(&Bs[buf][wn][0],      40, lane);
    bfr[1] = load_b_frag(&Bs[buf][wn + 16][0], 40, lane);
#pragma unroll
    for (int ti = 0; ti < 2; ++ti)
#pragma unroll
      for (int tj = 0; tj < 2; ++tj)
        acc[ti][tj] = wmma_bf16(af[ti], bfr[tj], acc[ti][tj]);
  }
#else
  // ---- synchronous fallback ----
  for (int k0 = 0; k0 < Kd; k0 += 32) {
    {
      const __bf16* g = gA + k0;
      *(v8bf*)&As[0][arow][akof]     = *(const v8bf*)g;
      *(v8bf*)&As[0][arow][akof + 8] = *(const v8bf*)(g + 8);
    }
    { *(v8bf*)&Bs[0][brow][bkof] = *(const v8bf*)(gB + k0); }
    __syncthreads();

    v16bf af[2], bfr[2];
    af[0]  = load_a_frag(&As[0][wm][0],      40, lane);
    af[1]  = load_a_frag(&As[0][wm + 16][0], 40, lane);
    bfr[0] = load_b_frag(&Bs[0][wn][0],      40, lane);
    bfr[1] = load_b_frag(&Bs[0][wn + 16][0], 40, lane);
#pragma unroll
    for (int ti = 0; ti < 2; ++ti)
#pragma unroll
      for (int tj = 0; tj < 2; ++tj)
        acc[ti][tj] = wmma_bf16(af[ti], bfr[tj], acc[ti][tj]);
    __syncthreads();
  }
#endif

  const int r = lane & 15, half = lane >> 4;
#pragma unroll
  for (int ti = 0; ti < 2; ++ti)
#pragma unroll
    for (int tj = 0; tj < 2; ++tj)
#pragma unroll
      for (int v = 0; v < 8; ++v) {
        const int row = m0 + wm + ti * 16 + v + 8 * half;
        const int col = n0 + wn + tj * 16 + r;
        C[(size_t)row * ldc + col] = acc[ti][tj][v];
      }
}

// ---------------- conv1d(K=4) + SiLU, split x / B / C ----------------
__global__ __launch_bounds__(256) void conv_silu_kernel(const float* __restrict__ zxbcdt,
                                                        const float* __restrict__ conv_w,
                                                        const float* __restrict__ conv_b,
                                                        float*  __restrict__ x_f32,
                                                        __bf16* __restrict__ x_bf16,
                                                        __bf16* __restrict__ B_bf16,
                                                        __bf16* __restrict__ C_bf16) {
  const size_t idx = (size_t)blockIdx.x * 256 + threadIdx.x;
  const size_t total = (size_t)kT * kCONVD;
  if (idx >= total) return;
  const int c = (int)(idx % kCONVD);
  const int t = (int)(idx / kCONVD);
  const int lb = t % kL;  // position inside batch (causal pad at batch start)
  float v = conv_b[c];
#pragma unroll
  for (int i = 0; i < kK; ++i) {
    const int li = lb - (kK - 1) + i;
    if (li >= 0)
      v += conv_w[c * kK + i] * zxbcdt[(size_t)(t - (kK - 1) + i) * kPROJ + kINTER + c];
  }
  const float s = v / (1.f + __expf(-v));  // silu
  if (c < kINTER) {
    x_f32[(size_t)t * kINTER + c]  = s;
    x_bf16[(size_t)t * kINTER + c] = f2bf(s);
  } else if (c < kINTER + kGN) {
    B_bf16[(size_t)t * kGN + (c - kINTER)] = f2bf(s);
  } else {
    C_bf16[(size_t)t * kGN + (c - kINTER - kGN)] = f2bf(s);
  }
}

// ---------------- dt softplus + per-chunk cumsum of dt*A ----------------
// block = (b, chunk, head), 256 threads = chunk positions
__global__ __launch_bounds__(256) void dt_scan_kernel(const float* __restrict__ zxbcdt,
                                                      const float* __restrict__ dt_bias,
                                                      const float* __restrict__ A_log,
                                                      float* __restrict__ dt_buf,
                                                      float* __restrict__ dAcs_buf,
                                                      float* __restrict__ w_buf,
                                                      float* __restrict__ decay_buf) {
  const int h  = blockIdx.x % kH;
  const int zc = (blockIdx.x / kH) % kNC;
  const int bb = blockIdx.x / (kH * kNC);
  const int j  = threadIdx.x;
  const int t  = (bb * kNC + zc) * kCS + j;

  float dtr = zxbcdt[(size_t)t * kPROJ + kINTER + kCONVD + h] + dt_bias[h];
  float dt  = (dtr > 20.f) ? dtr : log1pf(__expf(dtr));
  const float Ah = -__expf(A_log[h]);

  __shared__ float s[kCS];
  s[j] = dt * Ah;
  __syncthreads();
#pragma unroll
  for (int off = 1; off < kCS; off <<= 1) {
    float tv = (j >= off) ? s[j - off] : 0.f;
    __syncthreads();
    s[j] += tv;
    __syncthreads();
  }
  const float cs   = s[j];
  const float last = s[kCS - 1];
  dAcs_buf[(size_t)t * kH + h] = cs;
  dt_buf[(size_t)t * kH + h]   = dt;
  w_buf[(size_t)t * kH + h]    = __expf(last - cs) * dt;
  if (j == kCS - 1) decay_buf[(size_t)(bb * kNC + zc) * kH + h] = __expf(last);
}

// ---------------- per-chunk states: states[p,n] = sum_j (w_j*x_jp) * B_jn ----
// block = (b, chunk, head); WMMA GEMM M=P=64, N=kN=128, K=CS=256 (two K halves)
__global__ __launch_bounds__(256) void states_kernel(const __bf16* __restrict__ x_bf16,
                                                     const __bf16* __restrict__ B_bf16,
                                                     const float*  __restrict__ w_buf,
                                                     float* __restrict__ states) {
  __shared__ __align__(16) __bf16 xwT[64][136];   // [p][j-local]
  __shared__ __align__(16) __bf16 BT[128][136];   // [n][j-local]
  const int h  = blockIdx.x % kH;
  const int zc = (blockIdx.x / kH) % kNC;
  const int bb = blockIdx.x / (kH * kNC);
  const int g  = h >> 3;
  const int t0 = (bb * kNC + zc) * kCS;
  const int tid = threadIdx.x, lane = tid & 31, wid = tid >> 5;
  const int wm = (wid & 3) * 16;      // P rows
  const int wn = (wid >> 2) * 64;     // N cols

  v8f acc[4];
#pragma unroll
  for (int i = 0; i < 4; ++i) acc[i] = vzero8();

  for (int jh = 0; jh < 2; ++jh) {
    const int jb = jh * 128;
    for (int e = tid; e < 64 * 128; e += 256) {     // stage (w*x)^T
      const int p = e & 63, jj = e >> 6;
      const int t = t0 + jb + jj;
      const float xv = bf2f(x_bf16[(size_t)t * kINTER + h * kP + p]);
      xwT[p][jj] = f2bf(xv * w_buf[(size_t)t * kH + h]);
    }
    for (int e = tid; e < 128 * 128; e += 256) {    // stage B^T
      const int n = e & 127, jj = e >> 7;
      const int t = t0 + jb + jj;
      BT[n][jj] = B_bf16[(size_t)t * kGN + g * kN + n];
    }
    __syncthreads();
#pragma unroll
    for (int kk = 0; kk < 128; kk += 32) {
      v16bf af = load_a_frag(&xwT[wm][kk], 136, lane);
#pragma unroll
      for (int tj = 0; tj < 4; ++tj) {
        v16bf bfr = load_b_frag(&BT[wn + tj * 16][kk], 136, lane);
        acc[tj] = wmma_bf16(af, bfr, acc[tj]);
      }
    }
    __syncthreads();
  }

  const int r = lane & 15, half = lane >> 4;
  const size_t base = (((size_t)(bb * kNC + zc) * kH + h) * kP) * kN;
#pragma unroll
  for (int tj = 0; tj < 4; ++tj)
#pragma unroll
    for (int v = 0; v < 8; ++v) {
      const int p = wm + v + 8 * half;
      const int n = wn + tj * 16 + r;
      states[base + (size_t)p * kN + n] = acc[tj][v];
    }
}

// ---------------- sequential inter-chunk scan -> prev states (bf16) --------
__global__ __launch_bounds__(256) void scan_kernel(const float* __restrict__ states,
                                                   const float* __restrict__ decay_buf,
                                                   __bf16* __restrict__ prevb) {
  const int h  = blockIdx.x % kH;
  const int bb = blockIdx.x / kH;
  for (int e = threadIdx.x; e < kP * kN; e += 256) {
    float S = 0.f;
    for (int z = 0; z < kNC; ++z) {
      const size_t off = (((size_t)(bb * kNC + z) * kH + h) * kP * kN) + e;
      prevb[off] = f2bf(S);
      S = decay_buf[(size_t)(bb * kNC + z) * kH + h] * S + states[off];
    }
  }
}

// ---------------- fused chunk Y kernel ----------------
// block = (b, chunk, head); wave w owns rows i in [32w, 32w+32)
// Y = (mask(L).CB.dt) @ x  +  exp(dAcs) * (C @ prev^T)  +  D*x
__global__ __launch_bounds__(256) void chunk_y_kernel(const __bf16* __restrict__ Cb,
                                                      const __bf16* __restrict__ Bb,
                                                      const __bf16* __restrict__ xb,
                                                      const float*  __restrict__ x_f32,
                                                      const float*  __restrict__ dAcs_buf,
                                                      const float*  __restrict__ dt_buf,
                                                      const __bf16* __restrict__ prevb,
                                                      const float*  __restrict__ Dv,
                                                      float* __restrict__ Y) {
  __shared__ float dAcs_s[kCS];
  __shared__ float dt_s[kCS];
  __shared__ __align__(16) __bf16 xT[64][264];       // [p][j]
  __shared__ __align__(16) __bf16 Ml[8][32][40];     // per-wave M tile (bf16)

  const int h  = blockIdx.x % kH;
  const int zc = (blockIdx.x / kH) % kNC;
  const int bb = blockIdx.x / (kH * kNC);
  const int g  = h >> 3;
  const int t0 = (bb * kNC + zc) * kCS;
  const int tid = threadIdx.x, lane = tid & 31, wid = tid >> 5;
  const int r = lane & 15, half = lane >> 4;

  {
    const int j = tid;
    dAcs_s[j] = dAcs_buf[(size_t)(t0 + j) * kH + h];
    dt_s[j]   = dt_buf[(size_t)(t0 + j) * kH + h];
  }
  for (int e = tid; e < 64 * 256; e += 256) {        // stage x^T for this head
    const int p = e & 63, jj = e >> 6;
    xT[p][jj] = xb[(size_t)(t0 + jj) * kINTER + h * kP + p];
  }
  __syncthreads();

  const int i0 = wid * 32;
  v8f yacc[2][4];
#pragma unroll
  for (int a = 0; a < 2; ++a)
#pragma unroll
    for (int b = 0; b < 4; ++b) yacc[a][b] = vzero8();

  // causal j-blocks only (wave-uniform bound keeps EXEC all-ones)
  for (int jb = 0; jb <= wid; ++jb) {
    const int j0 = jb * 32;
    v8f cb[2][2];
#pragma unroll
    for (int a = 0; a < 2; ++a)
#pragma unroll
      for (int b = 0; b < 2; ++b) cb[a][b] = vzero8();

#pragma unroll
    for (int kk = 0; kk < kN; kk += 32) {            // CB = C @ B^T over n
      v16bf a0 = load_a_frag(Cb + (size_t)(t0 + i0)      * kGN + g * kN + kk, kGN, lane);
      v16bf a1 = load_a_frag(Cb + (size_t)(t0 + i0 + 16) * kGN + g * kN + kk, kGN, lane);
      v16bf b0 = load_b_frag(Bb + (size_t)(t0 + j0)      * kGN + g * kN + kk, kGN, lane);
      v16bf b1 = load_b_frag(Bb + (size_t)(t0 + j0 + 16) * kGN + g * kN + kk, kGN, lane);
      cb[0][0] = wmma_bf16(a0, b0, cb[0][0]);
      cb[0][1] = wmma_bf16(a0, b1, cb[0][1]);
      cb[1][0] = wmma_bf16(a1, b0, cb[1][0]);
      cb[1][1] = wmma_bf16(a1, b1, cb[1][1]);
    }

    // mask + decay + dt scale; round-trip through wave-private LDS to reshape
#pragma unroll
    for (int ti = 0; ti < 2; ++ti)
#pragma unroll
      for (int tj = 0; tj < 2; ++tj)
#pragma unroll
        for (int v = 0; v < 8; ++v) {
          const int li = ti * 16 + v + 8 * half;
          const int lj = tj * 16 + r;
          const int i = i0 + li, j = j0 + lj;
          const float val = (j <= i)
              ? cb[ti][tj][v] * __expf(dAcs_s[i] - dAcs_s[j]) * dt_s[j]
              : 0.f;
          Ml[wid][li][lj] = f2bf(val);
        }
    // same-wave LDS ops complete in order; fence the compiler only
    asm volatile("" ::: "memory");

#pragma unroll
    for (int ti = 0; ti < 2; ++ti) {
      v16bf am = load_a_frag(&Ml[wid][ti * 16][0], 40, lane);
#pragma unroll
      for (int tp = 0; tp < 4; ++tp) {
        v16bf bx = load_b_frag(&xT[tp * 16][j0], 264, lane);
        yacc[ti][tp] = wmma_bf16(am, bx, yacc[ti][tp]);
      }
    }
  }

  // Y_off = (C @ prev^T) * exp(dAcs_i)
  v8f off[2][4];
#pragma unroll
  for (int a = 0; a < 2; ++a)
#pragma unroll
    for (int b = 0; b < 4; ++b) off[a][b] = vzero8();

  const size_t pbase = ((size_t)(bb * kNC + zc) * kH + h) * kP * kN;
#pragma unroll
  for (int kk = 0; kk < kN; kk += 32) {
    v16bf a0 = load_a_frag(Cb + (size_t)(t0 + i0)      * kGN + g * kN + kk, kGN, lane);
    v16bf a1 = load_a_frag(Cb + (size_t)(t0 + i0 + 16) * kGN + g * kN + kk, kGN, lane);
#pragma unroll
    for (int tp = 0; tp < 4; ++tp) {
      v16bf bp = load_b_frag(prevb + pbase + (size_t)(tp * 16) * kN + kk, kN, lane);
      off[0][tp] = wmma_bf16(a0, bp, off[0][tp]);
      off[1][tp] = wmma_bf16(a1, bp, off[1][tp]);
    }
  }

  const float Dh = Dv[h];
#pragma unroll
  for (int ti = 0; ti < 2; ++ti)
#pragma unroll
    for (int tp = 0; tp < 4; ++tp)
#pragma unroll
      for (int v = 0; v < 8; ++v) {
        const int li = ti * 16 + v + 8 * half;
        const int i = i0 + li;
        const int t = t0 + i;
        const int p = tp * 16 + r;
        const float ev = __expf(dAcs_s[i]);
        const float xv = x_f32[(size_t)t * kINTER + h * kP + p];
        Y[(size_t)t * kINTER + h * kP + p] =
            yacc[ti][tp][v] + off[ti][tp][v] * ev + Dh * xv;
      }
}

// ---------------- RMSNorm * silu(z) gating -> bf16 ----------------
__global__ __launch_bounds__(256) void gate_norm_kernel(const float* __restrict__ Y,
                                                        const float* __restrict__ zxbcdt,
                                                        const float* __restrict__ norm_w,
                                                        __bf16* __restrict__ yb) {
  const int t = blockIdx.x;
  const int tid = threadIdx.x;
  __shared__ float red[256];
  float ss = 0.f;
  for (int c = tid; c < kINTER; c += 256) {
    const float y = Y[(size_t)t * kINTER + c];
    ss += y * y;
  }
  red[tid] = ss;
  __syncthreads();
#pragma unroll
  for (int s = 128; s > 0; s >>= 1) {
    if (tid < s) red[tid] += red[tid + s];
    __syncthreads();
  }
  const float inv = rsqrtf(red[0] / (float)kINTER + kEPS);
  for (int c = tid; c < kINTER; c += 256) {
    const float y = Y[(size_t)t * kINTER + c] * inv * norm_w[c];
    const float z = zxbcdt[(size_t)t * kPROJ + c];
    const float sz = z / (1.f + __expf(-z));
    yb[(size_t)t * kINTER + c] = f2bf(y * sz);
  }
}

// ---------------- host launcher ----------------
extern "C" void kernel_launch(void* const* d_in, const int* in_sizes, int n_in,
                              void* d_out, int out_size, void* d_ws, size_t ws_size,
                              hipStream_t stream) {
  const float* hs       = (const float*)d_in[0];   // [T, HID]
  const float* in_w     = (const float*)d_in[1];   // [PROJ, HID]
  const float* conv_w   = (const float*)d_in[2];   // [CONVD, K]
  const float* conv_b   = (const float*)d_in[3];   // [CONVD]
  const float* dt_bias  = (const float*)d_in[4];   // [H]
  const float* A_log    = (const float*)d_in[5];   // [H]
  const float* Dv       = (const float*)d_in[6];   // [H]
  const float* norm_w   = (const float*)d_in[7];   // [INTER]
  const float* out_w    = (const float*)d_in[8];   // [HID, INTER]
  float* out = (float*)d_out;                      // [T, HID]

  // workspace carve-up
  char* ws = (char*)d_ws;
  size_t off = 0;
  auto carve = [&](size_t bytes) -> char* {
    char* p = ws + off;
    off += (bytes + 255) & ~(size_t)255;
    return p;
  };
  __bf16* hsb     = (__bf16*)carve((size_t)kT * kHID * 2);
  __bf16* wib     = (__bf16*)carve((size_t)kPROJ * kHID * 2);
  __bf16* wob     = (__bf16*)carve((size_t)kHID * kINTER * 2);
  float*  zxbcdt  = (float*) carve((size_t)kT * kPROJ * 4);
  float*  x_f32   = (float*) carve((size_t)kT * kINTER * 4);
  __bf16* x_bf16  = (__bf16*)carve((size_t)kT * kINTER * 2);
  __bf16* B_bf16  = (__bf16*)carve((size_t)kT * kGN * 2);
  __bf16* C_bf16  = (__bf16*)carve((size_t)kT * kGN * 2);
  float*  dt_buf  = (float*) carve((size_t)kT * kH * 4);
  float*  dAcs    = (float*) carve((size_t)kT * kH * 4);
  float*  w_buf   = (float*) carve((size_t)kT * kH * 4);
  float*  decay   = (float*) carve((size_t)kB * kNC * kH * 4);
  float*  states  = (float*) carve((size_t)kB * kNC * kH * kP * kN * 4);
  __bf16* prevb   = (__bf16*)carve((size_t)kB * kNC * kH * kP * kN * 2);
  float*  Ybuf    = (float*) carve((size_t)kT * kINTER * 4);
  __bf16* yb      = (__bf16*)carve((size_t)kT * kINTER * 2);
  (void)ws_size; (void)n_in; (void)in_sizes; (void)out_size;

  // 1) fp32 -> bf16 converts
  cvt_bf16_kernel<<<2048, 256, 0, stream>>>(hs,    hsb, (size_t)kT * kHID);
  cvt_bf16_kernel<<<4096, 256, 0, stream>>>(in_w,  wib, (size_t)kPROJ * kHID);
  cvt_bf16_kernel<<<2048, 256, 0, stream>>>(out_w, wob, (size_t)kHID * kINTER);

  // 2) in_proj: zxbcdt[T, PROJ] = hs @ in_w^T
  gemm_bf16_nt<<<dim3(kPROJ / 64, kT / 128), 256, 0, stream>>>(
      hsb, wib, zxbcdt, kT, kPROJ, kHID, kHID, kHID, kPROJ);

  // 3) conv + silu
  {
    const size_t total = (size_t)kT * kCONVD;
    conv_silu_kernel<<<(unsigned)((total + 255) / 256), 256, 0, stream>>>(
        zxbcdt, conv_w, conv_b, x_f32, x_bf16, B_bf16, C_bf16);
  }

  // 4) dt softplus + cumsum(dt*A) per chunk
  dt_scan_kernel<<<kB * kNC * kH, 256, 0, stream>>>(
      zxbcdt, dt_bias, A_log, dt_buf, dAcs, w_buf, decay);

  // 5) per-chunk states (WMMA)
  states_kernel<<<kB * kNC * kH, 256, 0, stream>>>(x_bf16, B_bf16, w_buf, states);

  // 6) inter-chunk sequential scan -> prev (bf16)
  scan_kernel<<<kB * kH, 256, 0, stream>>>(states, decay, prevb);

  // 7) fused chunk Y (WMMA): Y_diag + Y_off + D*x
  chunk_y_kernel<<<kB * kNC * kH, 256, 0, stream>>>(
      C_bf16, B_bf16, x_bf16, x_f32, dAcs, dt_buf, prevb, Dv, Ybuf);

  // 8) RMSNorm gated -> bf16
  gate_norm_kernel<<<kT, 256, 0, stream>>>(Ybuf, zxbcdt, norm_w, yb);

  // 9) out_proj: out[T, HID] = yb @ out_w^T
  gemm_bf16_nt<<<dim3(kHID / 64, kT / 128), 256, 0, stream>>>(
      yb, wob, out, kT, kHID, kINTER, kINTER, kINTER, kHID);
}
